// EntropyGuidedAttention_74577812127912
// MI455X (gfx1250) — matrix operand order repaired and checked by reading, not compiled, not run on hardware
//
#include <hip/hip_runtime.h>
#include <hip/hip_bf16.h>

#define D_MODEL  512
#define N_HEADS  8
#define HEAD_DIM 64
#define T_SEQ    2048
#define B_SZ     2
#define TOPK     204          // max(1, int(2048*0.1))
#define ENT_THR  0.3f

typedef __attribute__((ext_vector_type(16))) __bf16 v16bf;
typedef __attribute__((ext_vector_type(8)))  float  v8f;

union ABu { unsigned int u[8]; v16bf v; };

#define WMMA_BF16(A, B, C) \
  __builtin_amdgcn_wmma_f32_16x16x32_bf16(false, (A), false, (B), (short)0, (C), false, false)

// ---------- helpers ----------

__device__ __forceinline__ unsigned short f2bf_bits(float f) {
  unsigned int u = __float_as_uint(f);
  u += 0x7FFFu + ((u >> 16) & 1u);   // round-to-nearest-even
  return (unsigned short)(u >> 16);
}

// A-matrix (16x32 bf16) per-lane fragment from row-major [M][K] storage.
__device__ __forceinline__ v16bf ldA(const unsigned short* row, int k0, int half) {
  ABu r;
  const unsigned int* p = (const unsigned int*)(row + k0);
  const int o = half ? 4 : 0;
#pragma unroll
  for (int v = 0; v < 4; ++v) { r.u[v] = p[o + v]; r.u[4 + v] = p[8 + o + v]; }
  return r.v;
}

// B-matrix (32x16 bf16) per-lane fragment from row-major [N][K] storage.
__device__ __forceinline__ v16bf ldB(const unsigned short* row, int k0, int half) {
  ABu r;
  const unsigned int* p = (const unsigned int*)(row + k0);
  const int o = half ? 4 : 0;
#pragma unroll
  for (int v = 0; v < 4; ++v) { r.u[v] = p[o + v]; r.u[4 + v] = p[8 + o + v]; }
  return r.v;
}

__device__ __forceinline__ float wave_max(float v) {
#pragma unroll
  for (int o = 16; o > 0; o >>= 1) v = fmaxf(v, __shfl_xor(v, o, 32));
  return v;
}
__device__ __forceinline__ int wave_sum_i(int v) {
#pragma unroll
  for (int o = 16; o > 0; o >>= 1) v += __shfl_xor(v, o, 32);
  return v;
}
__device__ __forceinline__ float wave_sum_f(float v) {
#pragma unroll
  for (int o = 16; o > 0; o >>= 1) v += __shfl_xor(v, o, 32);
  return v;
}

// ---------- kernel 1: fp32 -> bf16 cast ----------
__global__ void cast_bf16_kernel(const float* __restrict__ s,
                                 unsigned short* __restrict__ d, int n) {
  int i = blockIdx.x * 256 + threadIdx.x;
  if (i < n) d[i] = f2bf_bits(s[i]);
}

// ---------- kernel 2: entropy scores (conv1d + MLP + sigmoid), fp32 ----------
__global__ void entropy_kernel(const float* __restrict__ x,
                               const float* __restrict__ conv_w,
                               const float* __restrict__ conv_b,
                               const float* __restrict__ p1_w,
                               const float* __restrict__ p1_b,
                               const float* __restrict__ p2_w,
                               const float* __restrict__ p2_b,
                               float* __restrict__ ent) {
  extern __shared__ float sm[];
  float* xs   = sm;                // [40][512]
  float* feat = sm + 40 * D_MODEL; // [32][64]

  const int tid = threadIdx.x;
  const int b   = blockIdx.y;
  const int t0  = blockIdx.x * 32;

  for (int idx = tid; idx < 40 * D_MODEL; idx += 256) {
    int rr = idx >> 9, cc = idx & 511;
    int t = t0 + rr - 4;
    xs[idx] = (t >= 0 && t < T_SEQ) ? x[((size_t)b * T_SEQ + t) * D_MODEL + cc] : 0.f;
  }
  __syncthreads();

  for (int task = tid; task < 32 * 64; task += 256) {
    int pos = task >> 6, oc = task & 63;
    float acc = conv_b[oc];
    const float* cw = conv_w + (size_t)oc * D_MODEL * 8;
    for (int ic = 0; ic < D_MODEL; ++ic) {
      const float* cwi = cw + ic * 8;
#pragma unroll
      for (int w = 0; w < 8; ++w)
        acc = fmaf(xs[(pos + w) * D_MODEL + ic], cwi[w], acc);
    }
    feat[pos * 64 + oc] = acc;
  }
  __syncthreads();

  if (tid < 32) {
    float acc2 = p2_b[0];
    for (int j = 0; j < 32; ++j) {
      float hj = p1_b[j];
      const float* p1 = p1_w + j * 64;
#pragma unroll
      for (int i = 0; i < 64; ++i) hj = fmaf(feat[tid * 64 + i], p1[i], hj);
      hj = fmaxf(hj, 0.f);
      acc2 = fmaf(hj, p2_w[j], acc2);
    }
    ent[(size_t)b * T_SEQ + t0 + tid] = 1.f / (1.f + __expf(-acc2));
  }
}

// ---------- kernel 3: fused Q/K/V projection ----------
// grid: (256 M-tiles, 3 {q,k,v}), 256 threads (8 waves).
// Each wave: 4 col-tiles (64 cols), one shared A-fragment -> 4 WMMA per A-load.
// K-loop software-pipelined (next fragments loaded before current WMMAs).
__global__ void qkv_kernel(const unsigned short* __restrict__ xb,
                           const unsigned short* __restrict__ wqb,
                           const unsigned short* __restrict__ wkb,
                           const unsigned short* __restrict__ wvb,
                           const float* __restrict__ bq,
                           const float* __restrict__ bk,
                           const float* __restrict__ bv,
                           unsigned short* __restrict__ qo,
                           unsigned short* __restrict__ ko,
                           unsigned short* __restrict__ vto) {
  __shared__ unsigned short s_x[16][D_MODEL];   // 16 KB A tile

  const int tid = threadIdx.x, lane = tid & 31, wave = tid >> 5;
  const int half = lane >> 4, l16 = lane & 15;
  const int mt = blockIdx.x, z = blockIdx.y;
  const unsigned short* wmat = (z == 0) ? wqb : (z == 1) ? wkb : wvb;
  const float* bias = (z == 0) ? bq : (z == 1) ? bk : bv;

  {
    const unsigned int* src = (const unsigned int*)(xb + (size_t)mt * 16 * D_MODEL);
    unsigned int* dst = (unsigned int*)&s_x[0][0];
    for (int i = tid; i < 16 * D_MODEL / 2; i += 256) dst[i] = src[i];
  }
  __syncthreads();

  const int colbase = wave * 64;                 // 8 waves x 64 cols = 512
  const unsigned short* arow = &s_x[l16][0];
  const unsigned short* brow0 = wmat + (size_t)(colbase +  0 + l16) * D_MODEL;
  const unsigned short* brow1 = wmat + (size_t)(colbase + 16 + l16) * D_MODEL;
  const unsigned short* brow2 = wmat + (size_t)(colbase + 32 + l16) * D_MODEL;
  const unsigned short* brow3 = wmat + (size_t)(colbase + 48 + l16) * D_MODEL;

  v8f acc0 = {}, acc1 = {}, acc2 = {}, acc3 = {};
  v16bf a_c  = ldA(arow, 0, half);
  v16bf b0c = ldB(brow0, 0, half), b1c = ldB(brow1, 0, half);
  v16bf b2c = ldB(brow2, 0, half), b3c = ldB(brow3, 0, half);
#pragma unroll
  for (int ks = 0; ks < D_MODEL / 32 - 1; ++ks) {
    const int kn = (ks + 1) * 32;
    __builtin_prefetch(brow0 + kn + 256, 0, 1);
    v16bf a_n  = ldA(arow, kn, half);
    v16bf b0n = ldB(brow0, kn, half), b1n = ldB(brow1, kn, half);
    v16bf b2n = ldB(brow2, kn, half), b3n = ldB(brow3, kn, half);
    acc0 = WMMA_BF16(a_c, b0c, acc0);
    acc1 = WMMA_BF16(a_c, b1c, acc1);
    acc2 = WMMA_BF16(a_c, b2c, acc2);
    acc3 = WMMA_BF16(a_c, b3c, acc3);
    a_c = a_n; b0c = b0n; b1c = b1n; b2c = b2n; b3c = b3n;
  }
  acc0 = WMMA_BF16(a_c, b0c, acc0);
  acc1 = WMMA_BF16(a_c, b1c, acc1);
  acc2 = WMMA_BF16(a_c, b2c, acc2);
  acc3 = WMMA_BF16(a_c, b3c, acc3);

  v8f accs[4] = {acc0, acc1, acc2, acc3};
#pragma unroll
  for (int ti = 0; ti < 4; ++ti) {
    const int col = colbase + ti * 16 + l16;
    const float bval = bias[col];
    const int hh = col >> 6, dd = col & 63;
#pragma unroll
    for (int r = 0; r < 8; ++r) {
      int rowi = mt * 16 + (half ? 8 : 0) + r;     // rowi = b*T + t
      int b = rowi >> 11, t = rowi & (T_SEQ - 1);
      unsigned short val = f2bf_bits(accs[ti][r] + bval);
      size_t bh = (size_t)(b * N_HEADS + hh);
      if (z == 0)      qo [(bh * T_SEQ + t) * HEAD_DIM + dd] = val;
      else if (z == 1) ko [(bh * T_SEQ + t) * HEAD_DIM + dd] = val;
      else             vto[(bh * HEAD_DIM + dd) * T_SEQ + t] = val;  // v^T [d][t]
    }
  }
}

// ---------- kernel 4: fused attention: QK^T -> exact top-k -> softmax -> AV ----------
// grid: (T/16, H, B), 128 threads (4 waves).
// dyn LDS: scores 16x2048 f32 (128 KB, probs packed in-place) + q tile 2 KB
__global__ void attn_kernel(const unsigned short* __restrict__ qm,
                            const unsigned short* __restrict__ km,
                            const unsigned short* __restrict__ vtm,
                            unsigned short* __restrict__ ao) {
  extern __shared__ unsigned char smem[];
  float* s_scores = (float*)smem;                                   // [16][T_SEQ]
  unsigned short* s_q = (unsigned short*)(smem + 16 * T_SEQ * 4);   // [16][64]

  const int tid = threadIdx.x, lane = tid & 31, wave = tid >> 5;
  const int half = lane >> 4, l16 = lane & 15;
  const int qt = blockIdx.x, h = blockIdx.y, b = blockIdx.z;
  const int q0 = qt * 16;
  const size_t bh = (size_t)(b * N_HEADS + h);

  { // stage q tile (16 x 64 bf16)
    const unsigned int* src = (const unsigned int*)(qm + (bh * T_SEQ + q0) * HEAD_DIM);
    unsigned int* dst = (unsigned int*)s_q;
    for (int i = tid; i < 16 * HEAD_DIM / 2; i += 128) dst[i] = src[i];
  }
  __syncthreads();

  // QK^T: A fragments are loop-invariant; B double-buffered across key tiles.
  {
    const unsigned short* arow = s_q + l16 * HEAD_DIM;
    const v16bf aq0 = ldA(arow, 0, half);
    const v16bf aq1 = ldA(arow, 32, half);
    int nt = wave;
    const unsigned short* brow = km + (bh * T_SEQ + nt * 16 + l16) * HEAD_DIM;
    v16bf b0c = ldB(brow, 0, half), b1c = ldB(brow, 32, half);
#pragma unroll 4
    for (int i = 0; i < T_SEQ / 64 - 1; ++i) {     // 31 pipelined iterations
      const unsigned short* brn =
          km + (bh * T_SEQ + (nt + 4) * 16 + l16) * HEAD_DIM;
      __builtin_prefetch(brn + HEAD_DIM * 64, 0, 1);
      v16bf b0n = ldB(brn, 0, half), b1n = ldB(brn, 32, half);
      v8f acc = {};
      acc = WMMA_BF16(aq0, b0c, acc);
      acc = WMMA_BF16(aq1, b1c, acc);
#pragma unroll
      for (int r = 0; r < 8; ++r)
        s_scores[(size_t)((half ? 8 : 0) + r) * T_SEQ + nt * 16 + l16] =
            acc[r] * 0.125f;
      nt += 4; b0c = b0n; b1c = b1n;
    }
    v8f acc = {};
    acc = WMMA_BF16(aq0, b0c, acc);
    acc = WMMA_BF16(aq1, b1c, acc);
#pragma unroll
    for (int r = 0; r < 8; ++r)
      s_scores[(size_t)((half ? 8 : 0) + r) * T_SEQ + nt * 16 + l16] =
          acc[r] * 0.125f;
  }
  __syncthreads();

  // exact top-k threshold + softmax; each wave owns 4 rows
  for (int r = wave; r < 16; r += 4) {
    float* row = s_scores + (size_t)r * T_SEQ;
    unsigned int keys[64];
    float m = -3.4e38f;
#pragma unroll
    for (int j = 0; j < 64; ++j) {
      float f = row[j * 32 + lane];
      m = fmaxf(m, f);
      unsigned int u = __float_as_uint(f);
      keys[j] = (u & 0x80000000u) ? ~u : (u | 0x80000000u);   // monotone map
    }
    m = wave_max(m);

    unsigned int lo = 0u, hi = 0xFFFFFFFFu;   // k-th largest, bit-exact
    while (lo < hi) {
      unsigned int mid = lo + ((hi - lo + 1u) >> 1);
      int c = 0;
#pragma unroll
      for (int j = 0; j < 64; ++j) c += (keys[j] >= mid) ? 1 : 0;
      c = wave_sum_i(c);
      if (c >= TOPK) lo = mid; else hi = mid - 1u;
    }
    const float thr = (lo & 0x80000000u) ? __uint_as_float(lo ^ 0x80000000u)
                                         : __uint_as_float(~lo);

    float sum = 0.f;
#pragma unroll
    for (int j = 0; j < 64; ++j) {            // pass 1: read-only
      float f = row[j * 32 + lane];
      float p = (f >= thr) ? __expf(f - m) : 0.f;
      keys[j] = __float_as_uint(p);
      sum += p;
    }
    sum = wave_sum_f(sum);
    const float inv = 1.f / sum;

    unsigned short* prow = (unsigned short*)row;   // pack bf16 probs in place
#pragma unroll
    for (int j = 0; j < 64; ++j) {            // pass 2: write-only
      int idx = j * 32 + lane;
      prow[idx] = f2bf_bits(__uint_as_float(keys[j]) * inv);
    }
  }
  __syncthreads();

  // AV: wave w owns d-tile w; K-loop over all keys, software-pipelined.
  {
    const int d = wave * 16 + l16;
    const unsigned short* brow = vtm + (bh * HEAD_DIM + d) * T_SEQ;
    const unsigned short* arow = (const unsigned short*)(s_scores + (size_t)l16 * T_SEQ);
    v8f acc = {};
    v16bf a_c = ldA(arow, 0, half);
    v16bf b_c = ldB(brow, 0, half);
#pragma unroll 4
    for (int ks = 0; ks < T_SEQ / 32 - 1; ++ks) {  // 63 pipelined iterations
      const int kn = (ks + 1) * 32;
      __builtin_prefetch(brow + kn + 256, 0, 1);
      v16bf a_n = ldA(arow, kn, half);
      v16bf b_n = ldB(brow, kn, half);
      acc = WMMA_BF16(a_c, b_c, acc);
      a_c = a_n; b_c = b_n;
    }
    acc = WMMA_BF16(a_c, b_c, acc);
#pragma unroll
    for (int r = 0; r < 8; ++r) {
      int qrow = q0 + (half ? 8 : 0) + r;
      ao[((size_t)b * T_SEQ + qrow) * D_MODEL + h * HEAD_DIM + wave * 16 + l16] =
          f2bf_bits(acc[r]);
    }
  }
}

// ---------- kernel 5: output projection + entropy gating ----------
// grid: (256 M-tiles), 256 threads (8 waves), 4 col-tiles per wave, pipelined.
__global__ void oproj_kernel(const unsigned short* __restrict__ ab,
                             const unsigned short* __restrict__ wob,
                             const float* __restrict__ bo,
                             const float* __restrict__ x,
                             const float* __restrict__ ent,
                             float* __restrict__ out) {
  __shared__ unsigned short s_a[16][D_MODEL];
  const int tid = threadIdx.x, lane = tid & 31, wave = tid >> 5;
  const int half = lane >> 4, l16 = lane & 15;
  const int mt = blockIdx.x;

  {
    const unsigned int* src = (const unsigned int*)(ab + (size_t)mt * 16 * D_MODEL);
    unsigned int* dst = (unsigned int*)&s_a[0][0];
    for (int i = tid; i < 16 * D_MODEL / 2; i += 256) dst[i] = src[i];
  }
  __syncthreads();

  const int colbase = wave * 64;
  const unsigned short* arow = &s_a[l16][0];
  const unsigned short* brow0 = wob + (size_t)(colbase +  0 + l16) * D_MODEL;
  const unsigned short* brow1 = wob + (size_t)(colbase + 16 + l16) * D_MODEL;
  const unsigned short* brow2 = wob + (size_t)(colbase + 32 + l16) * D_MODEL;
  const unsigned short* brow3 = wob + (size_t)(colbase + 48 + l16) * D_MODEL;

  v8f acc0 = {}, acc1 = {}, acc2 = {}, acc3 = {};
  v16bf a_c  = ldA(arow, 0, half);
  v16bf b0c = ldB(brow0, 0, half), b1c = ldB(brow1, 0, half);
  v16bf b2c = ldB(brow2, 0, half), b3c = ldB(brow3, 0, half);
#pragma unroll
  for (int ks = 0; ks < D_MODEL / 32 - 1; ++ks) {
    const int kn = (ks + 1) * 32;
    __builtin_prefetch(brow0 + kn + 256, 0, 1);
    v16bf a_n  = ldA(arow, kn, half);
    v16bf b0n = ldB(brow0, kn, half), b1n = ldB(brow1, kn, half);
    v16bf b2n = ldB(brow2, kn, half), b3n = ldB(brow3, kn, half);
    acc0 = WMMA_BF16(a_c, b0c, acc0);
    acc1 = WMMA_BF16(a_c, b1c, acc1);
    acc2 = WMMA_BF16(a_c, b2c, acc2);
    acc3 = WMMA_BF16(a_c, b3c, acc3);
    a_c = a_n; b0c = b0n; b1c = b1n; b2c = b2n; b3c = b3n;
  }
  acc0 = WMMA_BF16(a_c, b0c, acc0);
  acc1 = WMMA_BF16(a_c, b1c, acc1);
  acc2 = WMMA_BF16(a_c, b2c, acc2);
  acc3 = WMMA_BF16(a_c, b3c, acc3);

  v8f accs[4] = {acc0, acc1, acc2, acc3};
#pragma unroll
  for (int ti = 0; ti < 4; ++ti) {
    const int col = colbase + ti * 16 + l16;
    const float bval = bo[col];
#pragma unroll
    for (int r = 0; r < 8; ++r) {
      int rowi = mt * 16 + (half ? 8 : 0) + r;   // rowi = b*T + t
      float y = accs[ti][r] + bval;
      bool hi = ent[rowi] > ENT_THR;
      size_t o = (size_t)rowi * D_MODEL + col;
      out[o] = hi ? y : x[o];
    }
  }
}

// ---------- launch ----------
extern "C" void kernel_launch(void* const* d_in, const int* in_sizes, int n_in,
                              void* d_out, int out_size, void* d_ws, size_t ws_size,
                              hipStream_t stream) {
  const float* x      = (const float*)d_in[0];
  const float* wq     = (const float*)d_in[1];
  const float* bq     = (const float*)d_in[2];
  const float* wk     = (const float*)d_in[3];
  const float* bk     = (const float*)d_in[4];
  const float* wv     = (const float*)d_in[5];
  const float* bv     = (const float*)d_in[6];
  const float* wo     = (const float*)d_in[7];
  const float* bo     = (const float*)d_in[8];
  const float* conv_w = (const float*)d_in[9];
  const float* conv_b = (const float*)d_in[10];
  const float* p1_w   = (const float*)d_in[11];
  const float* p1_b   = (const float*)d_in[12];
  const float* p2_w   = (const float*)d_in[13];
  const float* p2_b   = (const float*)d_in[14];

  float* out = (float*)d_out;                        // [B,T,512]
  float* ent = out + (size_t)B_SZ * T_SEQ * D_MODEL; // [B,T]

  const size_t NX = (size_t)B_SZ * T_SEQ * D_MODEL;  // 2,097,152
  const size_t NW = (size_t)D_MODEL * D_MODEL;       // 262,144

  unsigned short* xb  = (unsigned short*)d_ws;
  unsigned short* wqb = xb  + NX;
  unsigned short* wkb = wqb + NW;
  unsigned short* wvb = wkb + NW;
  unsigned short* wob = wvb + NW;
  unsigned short* qm  = wob + NW;   // [b][h][t][d]
  unsigned short* km  = qm  + NX;   // [b][h][t][d]
  unsigned short* vtm = km  + NX;   // [b][h][d][t]
  unsigned short* ab  = vtm + NX;   // attn out bf16 [b][t][c]

  cast_bf16_kernel<<<dim3((NX + 255) / 256), 256, 0, stream>>>(x,  xb,  (int)NX);
  cast_bf16_kernel<<<dim3((NW + 255) / 256), 256, 0, stream>>>(wq, wqb, (int)NW);
  cast_bf16_kernel<<<dim3((NW + 255) / 256), 256, 0, stream>>>(wk, wkb, (int)NW);
  cast_bf16_kernel<<<dim3((NW + 255) / 256), 256, 0, stream>>>(wv, wvb, (int)NW);
  cast_bf16_kernel<<<dim3((NW + 255) / 256), 256, 0, stream>>>(wo, wob, (int)NW);

  {
    size_t shm = (size_t)(40 * D_MODEL + 32 * 64) * sizeof(float);   // 90,112 B
    entropy_kernel<<<dim3(T_SEQ / 32, B_SZ), 256, shm, stream>>>(
        x, conv_w, conv_b, p1_w, p1_b, p2_w, p2_b, ent);
  }

  qkv_kernel<<<dim3(B_SZ * T_SEQ / 16, 3), 256, 0, stream>>>(
      xb, wqb, wkb, wvb, bq, bk, bv, qm, km, vtm);

  {
    size_t shm = (size_t)16 * T_SEQ * sizeof(float) + (size_t)16 * HEAD_DIM * 2; // 133,120 B
    attn_kernel<<<dim3(T_SEQ / 16, N_HEADS, B_SZ), 128, shm, stream>>>(
        qm, km, vtm, ab);
  }

  oproj_kernel<<<dim3(B_SZ * T_SEQ / 16), 256, 0, stream>>>(
      ab, wob, bo, x, ent, out);
}
